// TTWindowAttention_56281251447390
// MI455X (gfx1250) — matrix operand order repaired and checked by reading, not compiled, not run on hardware
//
#include <hip/hip_runtime.h>
#include <hip/hip_bf16.h>

typedef __bf16 bf16_t;
typedef __attribute__((ext_vector_type(16))) __bf16 v16bf;
typedef __attribute__((ext_vector_type(8)))  float  v8f;

#define NUM_HEADS 12
#define SEQ       49
#define HEAD_DIM  32
#define CDIM      384
#define BWIN      2048
#define NWIN      64

// ---------------- WMMA fragment helpers (wave32, ISA 7.12.2 layouts) -----------------
static __device__ __forceinline__ v16bf load_frag_a_lds(const bf16_t* base, int ld, int lane) {
  const int m  = lane & 15;
  const int hs = lane >> 4;
  v16bf a;
#pragma unroll
  for (int j = 0; j < 8; ++j) {
    const int kb = ((j & 4) ? 16 : 0) + hs * 8 + (j & 3) * 2;
    a[2 * j]     = base[m * ld + kb];
    a[2 * j + 1] = base[m * ld + kb + 1];
  }
  return a;
}

// B fragment from K-contiguous storage: element (k,n) = base[n*ld + k]
static __device__ __forceinline__ v16bf load_frag_bt_lds(const bf16_t* base, int ld, int lane) {
  const int n  = lane & 15;
  const int hs = lane >> 4;
  v16bf b;
#pragma unroll
  for (int j = 0; j < 8; ++j) {
    const int kb = ((j & 4) ? 16 : 0) + hs * 8 + (j & 3) * 2;
    b[2 * j]     = base[n * ld + kb];
    b[2 * j + 1] = base[n * ld + kb + 1];
  }
  return b;
}

static __device__ __forceinline__ v8f wmma_bf16(v16bf a, v16bf b, v8f c) {
  return __builtin_amdgcn_wmma_f32_16x16x32_bf16(false, a, false, b, (short)0, c, false, false);
}

#define LDA_PAD 34
#define LDBT    34

// ============================ Kernel 1: fused QKV projection =========================
// Block tile 128x64, 4 waves, each wave owns 32x64 (2 A frags x 4 B frags = 8 WMMA/step).
// Register-staged pipeline: next K-step's global loads issue while WMMAs run.
__global__ __launch_bounds__(128) void qkv_gemm_kernel(
    const float* __restrict__ X, const float* __restrict__ W,
    const float* __restrict__ bias, bf16_t* __restrict__ qkv) {
  __shared__ __align__(16) bf16_t As[128 * LDA_PAD];
  __shared__ __align__(16) bf16_t Bs[64 * LDBT];   // transposed: [n][k]

  const int tid  = threadIdx.x;
  const int lane = tid & 31;
  const int wv   = tid >> 5;
  const int n0   = blockIdx.x * 64;
  const int m0   = blockIdx.y * 128;
  const int K    = CDIM;
  const int NB   = 3 * CDIM;
  const float scale = 0.17677669529663687f;  // 32^-0.5

  const v8f vzero = {0.f, 0.f, 0.f, 0.f, 0.f, 0.f, 0.f, 0.f};
  v8f acc[2][4];
#pragma unroll
  for (int rr = 0; rr < 2; ++rr)
#pragma unroll
    for (int t = 0; t < 4; ++t) acc[rr][t] = vzero;

  float4 ra[8], rb[4];
  {  // prologue: stage K-step 0 into registers
    const float* Ag = X + (size_t)m0 * K;
#pragma unroll
    for (int i = 0; i < 8; ++i) {
      const int e = (tid + i * 128) * 4;
      ra[i] = *(const float4*)(Ag + (size_t)(e >> 5) * K + (e & 31));
    }
    const float* Bg = W + n0;
#pragma unroll
    for (int i = 0; i < 4; ++i) {
      const int e = (tid + i * 128) * 4;
      rb[i] = *(const float4*)(Bg + (size_t)(e >> 6) * NB + (e & 63));
    }
  }

  for (int k0 = 0; k0 < K; k0 += 32) {
    // dump staged registers to LDS (bf16 convert)
#pragma unroll
    for (int i = 0; i < 8; ++i) {
      const int e = (tid + i * 128) * 4;
      bf16_t* d = &As[(e >> 5) * LDA_PAD + (e & 31)];
      d[0] = (bf16_t)ra[i].x; d[1] = (bf16_t)ra[i].y;
      d[2] = (bf16_t)ra[i].z; d[3] = (bf16_t)ra[i].w;
    }
#pragma unroll
    for (int i = 0; i < 4; ++i) {
      const int e = (tid + i * 128) * 4;
      const int r = e >> 6, c = e & 63;
      Bs[(c + 0) * LDBT + r] = (bf16_t)rb[i].x;
      Bs[(c + 1) * LDBT + r] = (bf16_t)rb[i].y;
      Bs[(c + 2) * LDBT + r] = (bf16_t)rb[i].z;
      Bs[(c + 3) * LDBT + r] = (bf16_t)rb[i].w;
    }
    __syncthreads();

    // issue next K-step's global loads; they retire under the WMMAs below
    if (k0 + 32 < K) {
      const float* Ag = X + (size_t)m0 * K + (k0 + 32);
#pragma unroll
      for (int i = 0; i < 8; ++i) {
        const int e = (tid + i * 128) * 4;
        ra[i] = *(const float4*)(Ag + (size_t)(e >> 5) * K + (e & 31));
      }
      const float* Bg = W + (size_t)(k0 + 32) * NB + n0;
#pragma unroll
      for (int i = 0; i < 4; ++i) {
        const int e = (tid + i * 128) * 4;
        rb[i] = *(const float4*)(Bg + (size_t)(e >> 6) * NB + (e & 63));
      }
    }

    const v16bf a0 = load_frag_a_lds(&As[(wv * 32) * LDA_PAD], LDA_PAD, lane);
    const v16bf a1 = load_frag_a_lds(&As[(wv * 32 + 16) * LDA_PAD], LDA_PAD, lane);
#pragma unroll
    for (int t = 0; t < 4; ++t) {
      const v16bf b = load_frag_bt_lds(&Bs[(t * 16) * LDBT], LDBT, lane);
      acc[0][t] = wmma_bf16(a0, b, acc[0][t]);
      acc[1][t] = wmma_bf16(a1, b, acc[1][t]);
    }
    __syncthreads();
  }

  // Epilogue: +bias, scale q, scatter to (which, b, h, n, d) bf16 (incremental /49)
  const size_t SEGSZ = (size_t)BWIN * NUM_HEADS * SEQ * HEAD_DIM;
  const int ncol_base = n0 + (lane & 15);
#pragma unroll
  for (int rr = 0; rr < 2; ++rr) {
    const int mrow_base = m0 + wv * 32 + rr * 16 + ((lane >> 4) << 3);
    const int bb0 = mrow_base / SEQ;
    const int nn0 = mrow_base - bb0 * SEQ;
#pragma unroll
    for (int t = 0; t < 4; ++t) {
      const int col   = ncol_base + t * 16;
      const float bv  = bias[col];
      const int which = col / CDIM;
      const int rem   = col - which * CDIM;
      const int h     = rem >> 5;
      const int dd    = rem & 31;
      const float mul = (which == 0) ? scale : 1.0f;
      int bb = bb0, nn = nn0;
#pragma unroll
      for (int r = 0; r < 8; ++r) {
        const float v = (acc[rr][t][r] + bv) * mul;
        const size_t off = (size_t)which * SEGSZ +
                           (((size_t)bb * NUM_HEADS + h) * SEQ + nn) * HEAD_DIM + dd;
        qkv[off] = (bf16_t)v;
        if (++nn == SEQ) { nn = 0; ++bb; }
      }
    }
  }
}

// ======================= Kernel 2: per-(window,head) attention =======================
// q/k staged with GLOBAL_LOAD_ASYNC_TO_LDS_B128 (ASYNCcnt path); v staged transposed.
__global__ __launch_bounds__(128) void attn_kernel(
    const bf16_t* __restrict__ qkv, const float* __restrict__ rpb,
    const float* __restrict__ mask, float* __restrict__ attn_out) {
  __shared__ __align__(16) bf16_t q_s[64 * 32];
  __shared__ __align__(16) bf16_t k_s[64 * 32];
  __shared__ __align__(16) bf16_t vT[32 * 64];      // [d][token]
  __shared__ __align__(16) float  S_s[64 * 64];
  __shared__ __align__(16) bf16_t P_s[64 * 64];

  const int tid  = threadIdx.x;
  const int lane = tid & 31;
  const int wv   = tid >> 5;
  const int bh   = blockIdx.x;          // b*H + h
  const int b    = bh / NUM_HEADS;
  const int h    = bh - b * NUM_HEADS;
  const int widx = b & (NWIN - 1);      // window index = b % nW

  const size_t SEGSZ = (size_t)BWIN * NUM_HEADS * SEQ * HEAD_DIM;
  const size_t base  = (size_t)bh * (SEQ * HEAD_DIM);

  {
    // zero the pad region of q/k (uint2 chunks 392..511 = tokens 49..63)
    const uint2 z = {0u, 0u};
    uint2* qd = (uint2*)q_s; uint2* kd = (uint2*)k_s;
#pragma unroll
    for (int i = 0; i < 4; ++i) {
      const int c = tid + i * 128;
      if (c >= 392) { qd[c] = z; kd[c] = z; }
    }
    // async DMA of the valid 49x32 bf16 (3136 B = 196 x 16B chunks per tensor).
    // Per-lane: LDS dest offset in VDST, 64-bit global address in VADDR (GV mode).
    const unsigned q_lds = (unsigned)(uintptr_t)q_s;
    const unsigned k_lds = (unsigned)(uintptr_t)k_s;
    const char* qg = (const char*)(qkv + base);
    const char* kg = (const char*)(qkv + SEGSZ + base);
#pragma unroll
    for (int i = 0; i < 2; ++i) {
      const int c = tid + i * 128;      // 0..255
      if (c < 196) {
        const unsigned dq = q_lds + c * 16;
        const unsigned dk = k_lds + c * 16;
        const void* sq = qg + c * 16;
        const void* sk = kg + c * 16;
        asm volatile("global_load_async_to_lds_b128 %0, %1, off"
                     :: "v"(dq), "v"(sq) : "memory");
        asm volatile("global_load_async_to_lds_b128 %0, %1, off"
                     :: "v"(dk), "v"(sk) : "memory");
      }
    }
    // v: 49x32 token-major global -> 32x64 d-major LDS (zero pad tokens 49..63)
    const bf16_t* vg = qkv + 2 * SEGSZ + base;
#pragma unroll
    for (int i = 0; i < 4; ++i) {
      const int c = tid + i * 128;      // chunk: token = c>>3, d0 = (c&7)*4
      const int token = c >> 3;
      const int d0    = (c & 7) * 4;
      uint2 raw = {0u, 0u};
      if (c < 392) raw = *(const uint2*)(vg + (size_t)token * HEAD_DIM + d0);
      const bf16_t* e = (const bf16_t*)&raw;
      vT[(d0 + 0) * 64 + token] = e[0];
      vT[(d0 + 1) * 64 + token] = e[1];
      vT[(d0 + 2) * 64 + token] = e[2];
      vT[(d0 + 3) * 64 + token] = e[3];
    }
    asm volatile("s_wait_asynccnt 0x0" ::: "memory");
  }
  __syncthreads();

  {  // S = q @ k^T : wave wv owns score rows [16wv, 16wv+16)
    const v16bf a = load_frag_a_lds(&q_s[wv * 16 * 32], 32, lane);
    const int rbase = (wv * 16 + ((lane >> 4) << 3)) * 64;
    const int cbase = lane & 15;
    const v8f vzero = {0.f, 0.f, 0.f, 0.f, 0.f, 0.f, 0.f, 0.f};
#pragma unroll
    for (int t = 0; t < 4; ++t) {
      const v16bf bf = load_frag_bt_lds(&k_s[t * 16 * 32], 32, lane);
      const v8f c = wmma_bf16(a, bf, vzero);
#pragma unroll
      for (int r = 0; r < 8; ++r)
        S_s[rbase + r * 64 + t * 16 + cbase] = c[r];
    }
  }
  __syncthreads();

  // softmax over valid 49 cols, + rel_pos_bias + window mask; bf16 P with zero pad
  if (tid < 64) {
    const int row = tid;
    if (row < SEQ) {
      float* sr = &S_s[row * 64];
      const float* br = rpb  + ((size_t)h    * SEQ + row) * SEQ;
      const float* mr = mask + ((size_t)widx * SEQ + row) * SEQ;
      float mx = -3.4e38f;
      for (int c = 0; c < SEQ; ++c) {
        const float v = sr[c] + br[c] + mr[c];
        sr[c] = v;
        mx = fmaxf(mx, v);
      }
      float sum = 0.f;
      for (int c = 0; c < SEQ; ++c) {
        const float e = __expf(sr[c] - mx);
        sr[c] = e;
        sum += e;
      }
      const float inv = 1.0f / sum;
      for (int c = 0; c < SEQ; ++c) P_s[row * 64 + c] = (bf16_t)(sr[c] * inv);
      for (int c = SEQ; c < 64; ++c) P_s[row * 64 + c] = (bf16_t)0.f;
    } else {
      for (int c = 0; c < 64; ++c) P_s[row * 64 + c] = (bf16_t)0.f;
    }
  }
  __syncthreads();

  {  // O = P @ v
    const int rowtok_base = wv * 16 + ((lane >> 4) << 3);
    const int cb = lane & 15;
    const v8f vzero = {0.f, 0.f, 0.f, 0.f, 0.f, 0.f, 0.f, 0.f};
#pragma unroll
    for (int ct = 0; ct < 2; ++ct) {
      v8f acc = vzero;
#pragma unroll
      for (int kt = 0; kt < 2; ++kt) {
        const v16bf a = load_frag_a_lds(&P_s[wv * 16 * 64 + kt * 32], 64, lane);
        const v16bf bb = load_frag_bt_lds(&vT[(ct * 16) * 64 + kt * 32], 64, lane);
        acc = wmma_bf16(a, bb, acc);
      }
#pragma unroll
      for (int r = 0; r < 8; ++r) {
        const int tok = rowtok_base + r;
        if (tok < SEQ) {
          const size_t off = ((size_t)b * SEQ + tok) * CDIM + h * HEAD_DIM + ct * 16 + cb;
          attn_out[off] = acc[r];
        }
      }
    }
  }
}

// ============================ Kernel 3: output projection ============================
__global__ __launch_bounds__(128) void proj_gemm_kernel(
    const float* __restrict__ A, const float* __restrict__ W,
    const float* __restrict__ bias, float* __restrict__ out) {
  __shared__ __align__(16) bf16_t As[128 * LDA_PAD];
  __shared__ __align__(16) bf16_t Bs[64 * LDBT];   // transposed: [n][k]

  const int tid  = threadIdx.x;
  const int lane = tid & 31;
  const int wv   = tid >> 5;
  const int n0   = blockIdx.x * 64;
  const int m0   = blockIdx.y * 128;
  const int K    = CDIM;
  const int NB   = CDIM;

  const v8f vzero = {0.f, 0.f, 0.f, 0.f, 0.f, 0.f, 0.f, 0.f};
  v8f acc[2][4];
#pragma unroll
  for (int rr = 0; rr < 2; ++rr)
#pragma unroll
    for (int t = 0; t < 4; ++t) acc[rr][t] = vzero;

  float4 ra[8], rb[4];
  {
    const float* Ag = A + (size_t)m0 * K;
#pragma unroll
    for (int i = 0; i < 8; ++i) {
      const int e = (tid + i * 128) * 4;
      ra[i] = *(const float4*)(Ag + (size_t)(e >> 5) * K + (e & 31));
    }
    const float* Bg = W + n0;
#pragma unroll
    for (int i = 0; i < 4; ++i) {
      const int e = (tid + i * 128) * 4;
      rb[i] = *(const float4*)(Bg + (size_t)(e >> 6) * NB + (e & 63));
    }
  }

  for (int k0 = 0; k0 < K; k0 += 32) {
#pragma unroll
    for (int i = 0; i < 8; ++i) {
      const int e = (tid + i * 128) * 4;
      bf16_t* d = &As[(e >> 5) * LDA_PAD + (e & 31)];
      d[0] = (bf16_t)ra[i].x; d[1] = (bf16_t)ra[i].y;
      d[2] = (bf16_t)ra[i].z; d[3] = (bf16_t)ra[i].w;
    }
#pragma unroll
    for (int i = 0; i < 4; ++i) {
      const int e = (tid + i * 128) * 4;
      const int r = e >> 6, c = e & 63;
      Bs[(c + 0) * LDBT + r] = (bf16_t)rb[i].x;
      Bs[(c + 1) * LDBT + r] = (bf16_t)rb[i].y;
      Bs[(c + 2) * LDBT + r] = (bf16_t)rb[i].z;
      Bs[(c + 3) * LDBT + r] = (bf16_t)rb[i].w;
    }
    __syncthreads();

    if (k0 + 32 < K) {
      const float* Ag = A + (size_t)m0 * K + (k0 + 32);
#pragma unroll
      for (int i = 0; i < 8; ++i) {
        const int e = (tid + i * 128) * 4;
        ra[i] = *(const float4*)(Ag + (size_t)(e >> 5) * K + (e & 31));
      }
      const float* Bg = W + (size_t)(k0 + 32) * NB + n0;
#pragma unroll
      for (int i = 0; i < 4; ++i) {
        const int e = (tid + i * 128) * 4;
        rb[i] = *(const float4*)(Bg + (size_t)(e >> 6) * NB + (e & 63));
      }
    }

    const v16bf a0 = load_frag_a_lds(&As[(wv * 32) * LDA_PAD], LDA_PAD, lane);
    const v16bf a1 = load_frag_a_lds(&As[(wv * 32 + 16) * LDA_PAD], LDA_PAD, lane);
#pragma unroll
    for (int t = 0; t < 4; ++t) {
      const v16bf b = load_frag_bt_lds(&Bs[(t * 16) * LDBT], LDBT, lane);
      acc[0][t] = wmma_bf16(a0, b, acc[0][t]);
      acc[1][t] = wmma_bf16(a1, b, acc[1][t]);
    }
    __syncthreads();
  }

  const int ncol_base = n0 + (lane & 15);
#pragma unroll
  for (int rr = 0; rr < 2; ++rr) {
    const int mrow_base = m0 + wv * 32 + rr * 16 + ((lane >> 4) << 3);
#pragma unroll
    for (int t = 0; t < 4; ++t) {
      const int col  = ncol_base + t * 16;
      const float bv = bias[col];
#pragma unroll
      for (int r = 0; r < 8; ++r) {
        const int mg = mrow_base + r;
        out[(size_t)mg * CDIM + col] = acc[rr][t][r] + bv;
      }
    }
  }
}

// ======================================================================================
extern "C" void kernel_launch(void* const* d_in, const int* in_sizes, int n_in,
                              void* d_out, int out_size, void* d_ws, size_t ws_size,
                              hipStream_t stream) {
  (void)in_sizes; (void)n_in; (void)out_size; (void)ws_size;
  const float* x      = (const float*)d_in[0];
  const float* mask   = (const float*)d_in[1];
  const float* qkv_w  = (const float*)d_in[2];
  const float* qkv_b  = (const float*)d_in[3];
  const float* proj_w = (const float*)d_in[4];
  const float* proj_b = (const float*)d_in[5];
  const float* rpb    = (const float*)d_in[6];
  float* out = (float*)d_out;

  // workspace: [ qkv bf16 : 3 * 2048*12*49*32 elems ][ attn_out f32 : 100352*384 ]
  const size_t SEGSZ = (size_t)BWIN * NUM_HEADS * SEQ * HEAD_DIM;
  bf16_t* qkv_ws = (bf16_t*)d_ws;
  float*  attn_ws = (float*)((char*)d_ws + 3 * SEGSZ * sizeof(bf16_t));

  const dim3 blk(128);
  qkv_gemm_kernel<<<dim3(18, 784), blk, 0, stream>>>(x, qkv_w, qkv_b, qkv_ws);
  attn_kernel<<<dim3(BWIN * NUM_HEADS), blk, 0, stream>>>(qkv_ws, rpb, mask, attn_ws);
  proj_gemm_kernel<<<dim3(6, 784), blk, 0, stream>>>(attn_ws, proj_w, proj_b, out);
}
